// BiRNN_6399501271114
// MI455X (gfx1250) — compile-verified, matrix operand order checked by reading, not run on hardware
//
#include <hip/hip_runtime.h>
#include <hip/hip_bf16.h>

// ---------------------------------------------------------------------------
// Bidirectional LSTM for MI455X (gfx1250).
//  - bf16 WMMA (v_wmma_f32_16x16x32_bf16) for all matmuls, f32 accumulate.
//  - Recurrence: one persistent workgroup (16 waves), state in LDS
//    (z: 128KB f32, c: 32KB f32, h: 16KB bf16  -> 176KB of the 320KB WGP LDS).
//  - Weights transpose-converted to bf16 [N][K] so WMMA B-fragments are
//    contiguous 32B/lane loads (k = 16*(lane/16) + s, the same order as the A loads).
// ---------------------------------------------------------------------------

typedef __attribute__((ext_vector_type(16))) __bf16 v16bf;
typedef __attribute__((ext_vector_type(8)))  float  v8f;

#define WMMA_BF16(a, b, c)                                                     \
  __builtin_amdgcn_wmma_f32_16x16x32_bf16(false, (a), false, (b), (short)0,    \
                                          (c), false, false)

constexpr int Bn = 32;    // batch
constexpr int Tn = 4096;  // time steps
constexpr int Dn = 256;   // input feature dim
constexpr int Hn = 256;   // hidden dim
constexpr int Gn = 1024;  // 4*H gate dim
constexpr int On = 512;   // dense output dim
constexpr int KD = 512;   // dense K = 2*H

// ---------------------------------------------------------------------------
// Weight transpose + f32 -> bf16 convert:  src[K][N] (row major) -> dst[N][K]
// ---------------------------------------------------------------------------
__global__ void k_transpose_bf16(const float* __restrict__ src,
                                 __bf16* __restrict__ dst, int K, int N) {
  int total = K * N;
  for (int idx = blockIdx.x * blockDim.x + threadIdx.x; idx < total;
       idx += gridDim.x * blockDim.x) {
    int k = idx / N;
    int n = idx - k * N;
    dst[(size_t)n * K + k] = (__bf16)src[idx];
  }
}

// ---------------------------------------------------------------------------
// Input projection GEMM:  XP[r][g] = sum_k x_row(r)[k] * WxT[g][k]
// rows r = t*B + b  (r in [0, T*B)),  x is (B,T,D) so x_row = x[b][t][:].
// Block = 256 threads (8 waves), block tile 128x128, wave tile 32x64.
// ---------------------------------------------------------------------------
__global__ __launch_bounds__(256) void k_xproj(const float* __restrict__ x,
                                               const __bf16* __restrict__ WxT,
                                               float* __restrict__ xout) {
  const int lane = threadIdx.x & 31;
  const int wave = threadIdx.x >> 5;
  const int wm = wave & 3;   // 4 wave rows
  const int wn = wave >> 2;  // 2 wave cols
  const int rowBase = blockIdx.x * 128 + wm * 32;
  const int colBase = blockIdx.y * 128 + wn * 64;
  const int l16 = lane & 15;
  const int lhi = lane >> 4;

  const v8f vzero = {0.f, 0.f, 0.f, 0.f, 0.f, 0.f, 0.f, 0.f};
  v8f acc[2][4];
#pragma unroll
  for (int mt = 0; mt < 2; ++mt)
#pragma unroll
    for (int j = 0; j < 4; ++j) acc[mt][j] = vzero;

  const float* xr[2];
#pragma unroll
  for (int mt = 0; mt < 2; ++mt) {
    int r = rowBase + mt * 16 + l16;
    int b = r & (Bn - 1);
    int t = r >> 5;
    xr[mt] = x + ((size_t)b * Tn + t) * Dn + lhi * 16;
  }

  for (int k0 = 0; k0 < Dn; k0 += 32) {
    v16bf a[2];
#pragma unroll
    for (int mt = 0; mt < 2; ++mt) {
      const float* p = xr[mt] + k0;
      v16bf av;
#pragma unroll
      for (int e = 0; e < 16; ++e) av[e] = (__bf16)p[e];
      a[mt] = av;
    }
#pragma unroll
    for (int j = 0; j < 4; ++j) {
      const __bf16* bp =
          WxT + (size_t)(colBase + j * 16 + l16) * Dn + k0 + lhi * 16;
      v16bf bf = *(const v16bf*)bp;
      acc[0][j] = WMMA_BF16(a[0], bf, acc[0][j]);
      acc[1][j] = WMMA_BF16(a[1], bf, acc[1][j]);
    }
  }

#pragma unroll
  for (int mt = 0; mt < 2; ++mt)
#pragma unroll
    for (int j = 0; j < 4; ++j)
#pragma unroll
      for (int v = 0; v < 8; ++v) {
        int r = rowBase + mt * 16 + lhi * 8 + v;
        int c = colBase + j * 16 + l16;
        xout[(size_t)r * Gn + c] = acc[mt][j][v];
      }
}

// ---------------------------------------------------------------------------
// Persistent recurrent kernel: one workgroup (16 waves / 512 threads) runs
// the whole scan. LDS holds z (32x1024 f32), c (32x256 f32), h (32x256 bf16).
// Each step: z = h @ Wh (WMMA) -> barrier -> gate nonlinearity, update c/h,
// emit relu(h) as bf16 -> barrier.
// ---------------------------------------------------------------------------
__global__ __launch_bounds__(512) void k_lstm(
    const float* __restrict__ xproj,   // [T][B][G] precomputed x @ Wx
    const __bf16* __restrict__ WhT,    // [G][H] transposed recurrent weights
    const float* __restrict__ bias,    // [G]
    const float* __restrict__ cinit,   // [B][H]
    const float* __restrict__ hinit,   // [B][H]
    float* __restrict__ cfin,          // [B][H] final carry out
    float* __restrict__ hfin,          // [B][H]
    __bf16* __restrict__ hout,         // [T][B][H]  relu(h), bf16
    int reverse) {
  extern __shared__ char smem[];
  float* zs = (float*)smem;                                   // 32*1024 f32
  float* cs = (float*)(smem + (size_t)Bn * Gn * sizeof(float));  // 32*256 f32
  __bf16* hs =
      (__bf16*)(smem + (size_t)(Bn * Gn + Bn * Hn) * sizeof(float));  // bf16

  const int tid = threadIdx.x;
  const int lane = tid & 31;
  const int wave = tid >> 5;
  const int l16 = lane & 15;
  const int lhi = lane >> 4;
  const int n0base = wave * 64;  // 16 waves x 64 gate-columns = 1024

  for (int i = tid; i < Bn * Hn; i += 512) {
    cs[i] = cinit[i];
    hs[i] = (__bf16)hinit[i];
  }
  __syncthreads();

  const v8f vzero = {0.f, 0.f, 0.f, 0.f, 0.f, 0.f, 0.f, 0.f};

  for (int s = 0; s < Tn; ++s) {
    const int t = reverse ? (Tn - 1 - s) : s;

    // ---- GEMM phase: z = h(32x256) @ Wh(256x1024) ----
    v8f acc[2][4];
#pragma unroll
    for (int mt = 0; mt < 2; ++mt)
#pragma unroll
      for (int j = 0; j < 4; ++j) acc[mt][j] = vzero;

    for (int k0 = 0; k0 < Hn; k0 += 32) {
      v16bf a0 = *(const v16bf*)(hs + (size_t)l16 * Hn + k0 + lhi * 16);
      v16bf a1 = *(const v16bf*)(hs + (size_t)(16 + l16) * Hn + k0 + lhi * 16);
#pragma unroll
      for (int j = 0; j < 4; ++j) {
        const __bf16* bp =
            WhT + (size_t)(n0base + j * 16 + l16) * Hn + k0 + lhi * 16;
        v16bf bf = *(const v16bf*)bp;
        acc[0][j] = WMMA_BF16(a0, bf, acc[0][j]);
        acc[1][j] = WMMA_BF16(a1, bf, acc[1][j]);
      }
    }
#pragma unroll
    for (int mt = 0; mt < 2; ++mt)
#pragma unroll
      for (int j = 0; j < 4; ++j)
#pragma unroll
        for (int v = 0; v < 8; ++v)
          zs[(size_t)(mt * 16 + lhi * 8 + v) * Gn + n0base + j * 16 + l16] =
              acc[mt][j][v];
    __syncthreads();

    // ---- gate phase: i,f,g,o split along G; update c,h ----
    const float* xz = xproj + (size_t)t * Bn * Gn;
    for (int idx = tid; idx < Bn * Hn; idx += 512) {
      int b = idx >> 8;        // H = 256
      int j = idx & (Hn - 1);
      const float* zr = zs + (size_t)b * Gn;
      const float* xr = xz + (size_t)b * Gn;
      float zi = zr[j] + xr[j] + bias[j];
      float zf = zr[j + 256] + xr[j + 256] + bias[j + 256];
      float zg = zr[j + 512] + xr[j + 512] + bias[j + 512];
      float zo = zr[j + 768] + xr[j + 768] + bias[j + 768];
      float si = 1.f / (1.f + __expf(-zi));
      float sf = 1.f / (1.f + __expf(-zf));
      float so = 1.f / (1.f + __expf(-zo));
      float cn = sf * cs[idx] + si * tanhf(zg);
      float hn = so * tanhf(cn);
      cs[idx] = cn;
      hs[idx] = (__bf16)hn;
      hout[(size_t)t * Bn * Hn + idx] = (__bf16)fmaxf(hn, 0.f);
    }
    __syncthreads();
  }

  for (int i = tid; i < Bn * Hn; i += 512) {
    cfin[i] = cs[i];
    hfin[i] = (float)hs[i];
  }
}

// ---------------------------------------------------------------------------
// Dense output GEMM: out[b][t][o] = relu(hcat[t][b]) @ W_dense + b_dense
// K = 512 split over hF (k<256) and hB (k>=256); (T,B)->(B,T) fused in store.
// ---------------------------------------------------------------------------
__global__ __launch_bounds__(256) void k_dense(const __bf16* __restrict__ hF,
                                               const __bf16* __restrict__ hB,
                                               const __bf16* __restrict__ WdT,
                                               const float* __restrict__ bd,
                                               float* __restrict__ out) {
  const int lane = threadIdx.x & 31;
  const int wave = threadIdx.x >> 5;
  const int wm = wave & 3;
  const int wn = wave >> 2;
  const int rowBase = blockIdx.x * 128 + wm * 32;
  const int colBase = blockIdx.y * 128 + wn * 64;
  const int l16 = lane & 15;
  const int lhi = lane >> 4;

  const v8f vzero = {0.f, 0.f, 0.f, 0.f, 0.f, 0.f, 0.f, 0.f};
  v8f acc[2][4];
#pragma unroll
  for (int mt = 0; mt < 2; ++mt)
#pragma unroll
    for (int j = 0; j < 4; ++j) acc[mt][j] = vzero;

  const __bf16* aF[2];
  const __bf16* aB[2];
#pragma unroll
  for (int mt = 0; mt < 2; ++mt) {
    size_t r = (size_t)(rowBase + mt * 16 + l16);
    aF[mt] = hF + r * Hn + lhi * 16;
    aB[mt] = hB + r * Hn + lhi * 16;
  }

  for (int k0 = 0; k0 < KD; k0 += 32) {
    v16bf a[2];
#pragma unroll
    for (int mt = 0; mt < 2; ++mt) {
      const __bf16* p = (k0 < Hn) ? (aF[mt] + k0) : (aB[mt] + (k0 - Hn));
      a[mt] = *(const v16bf*)p;
    }
#pragma unroll
    for (int j = 0; j < 4; ++j) {
      const __bf16* bp =
          WdT + (size_t)(colBase + j * 16 + l16) * KD + k0 + lhi * 16;
      v16bf bf = *(const v16bf*)bp;
      acc[0][j] = WMMA_BF16(a[0], bf, acc[0][j]);
      acc[1][j] = WMMA_BF16(a[1], bf, acc[1][j]);
    }
  }

#pragma unroll
  for (int j = 0; j < 4; ++j) {
    int o = colBase + j * 16 + l16;
    float bias = bd[o];
#pragma unroll
    for (int mt = 0; mt < 2; ++mt)
#pragma unroll
      for (int v = 0; v < 8; ++v) {
        int r = rowBase + mt * 16 + lhi * 8 + v;  // r = t*B + b
        int b = r & (Bn - 1);
        int t = r >> 5;
        out[((size_t)b * Tn + t) * On + o] = acc[mt][j][v] + bias;
      }
  }
}

// ---------------------------------------------------------------------------
// Host side
// ---------------------------------------------------------------------------
extern "C" void kernel_launch(void* const* d_in, const int* in_sizes, int n_in,
                              void* d_out, int out_size, void* d_ws,
                              size_t ws_size, hipStream_t stream) {
  const float* carry_c = (const float*)d_in[0];
  const float* carry_h = (const float*)d_in[1];
  const float* x       = (const float*)d_in[2];
  const float* Wx_f    = (const float*)d_in[3];
  const float* Wh_f    = (const float*)d_in[4];
  const float* b_f     = (const float*)d_in[5];
  const float* Wx_b    = (const float*)d_in[6];
  const float* Wh_b    = (const float*)d_in[7];
  const float* b_b     = (const float*)d_in[8];
  const float* W_dense = (const float*)d_in[9];
  const float* b_dense = (const float*)d_in[10];
  float* out = (float*)d_out;

  char* ws = (char*)d_ws;
  size_t off = 0;
  auto alloc = [&](size_t bytes) -> void* {
    void* p = ws + off;
    off += (bytes + 255) & ~(size_t)255;
    return p;
  };

  float*  XF   = (float*)alloc((size_t)Tn * Bn * Gn * sizeof(float));  // 512MB
  float*  XB   = (float*)alloc((size_t)Tn * Bn * Gn * sizeof(float));  // 512MB
  __bf16* hF   = (__bf16*)alloc((size_t)Tn * Bn * Hn * sizeof(__bf16)); // 64MB
  __bf16* hB   = (__bf16*)alloc((size_t)Tn * Bn * Hn * sizeof(__bf16)); // 64MB
  __bf16* WxTf = (__bf16*)alloc((size_t)Gn * Dn * sizeof(__bf16));
  __bf16* WxTb = (__bf16*)alloc((size_t)Gn * Dn * sizeof(__bf16));
  __bf16* WhTf = (__bf16*)alloc((size_t)Gn * Hn * sizeof(__bf16));
  __bf16* WhTb = (__bf16*)alloc((size_t)Gn * Hn * sizeof(__bf16));
  __bf16* WdT  = (__bf16*)alloc((size_t)On * KD * sizeof(__bf16));
  float*  cfin = (float*)alloc((size_t)Bn * Hn * sizeof(float));
  float*  hfin = (float*)alloc((size_t)Bn * Hn * sizeof(float));

  // 1) weight transpose + bf16 convert
  k_transpose_bf16<<<1024, 256, 0, stream>>>(Wx_f, WxTf, Dn, Gn);
  k_transpose_bf16<<<1024, 256, 0, stream>>>(Wx_b, WxTb, Dn, Gn);
  k_transpose_bf16<<<1024, 256, 0, stream>>>(Wh_f, WhTf, Hn, Gn);
  k_transpose_bf16<<<1024, 256, 0, stream>>>(Wh_b, WhTb, Hn, Gn);
  k_transpose_bf16<<<1024, 256, 0, stream>>>(W_dense, WdT, KD, On);

  // 2) input projections (throughput GEMMs, full grid)
  dim3 gx((Tn * Bn) / 128, Gn / 128);
  k_xproj<<<gx, 256, 0, stream>>>(x, WxTf, XF);
  k_xproj<<<gx, 256, 0, stream>>>(x, WxTb, XB);

  // 3) recurrence (latency-bound, one persistent WGP each; backward consumes
  //    forward's final carry, so the two launches are stream-ordered)
  size_t lds_bytes =
      (size_t)(Bn * Gn + Bn * Hn) * sizeof(float) + (size_t)Bn * Hn * 2;
  k_lstm<<<1, 512, lds_bytes, stream>>>(XF, WhTf, b_f, carry_c, carry_h, cfin,
                                        hfin, hF, 0);
  k_lstm<<<1, 512, lds_bytes, stream>>>(XB, WhTb, b_b, cfin, hfin, cfin, hfin,
                                        hB, 1);

  // 4) dense output layer with fused (T,B)->(B,T) transpose
  dim3 gd((Tn * Bn) / 128, On / 128);
  k_dense<<<gd, 256, 0, stream>>>(hF, hB, WdT, b_dense, out);
}
